// SASRec_8100308320515
// MI455X (gfx1250) — compile-verified, matrix operand order checked by reading, not run on hardware
//
#include <hip/hip_runtime.h>
#include <hip/hip_fp16.h>

// ---------------------------------------------------------------------------
// SASRec + SparseMoE forward for MI455X (gfx1250), wave32 WMMA f16/f32-acc.
// All big GEMMs use v_wmma_f32_16x16x32_f16, fully unrolled over K=256.
// Weights are pre-converted (and experts pre-transposed) to f16 once per
// block so every B-fragment is two aligned b128 loads.
// Workspace budget: ~108 MB.
// ---------------------------------------------------------------------------

#define Hh   256
#define Ll   200
#define Bb   128
#define NBb  2
#define NHh  4
#define Ee   8
#define HDh  64
#define LPAD 224   // padded key length (multiple of 32, >= 208)

typedef __attribute__((ext_vector_type(16))) _Float16 vh16;
typedef __attribute__((ext_vector_type(8)))  _Float16 vh8;
typedef __attribute__((ext_vector_type(8)))  float    vf8;
typedef __attribute__((ext_vector_type(4)))  float    vf4;

#define WMMA16(a, b, c) \
  __builtin_amdgcn_wmma_f32_16x16x32_f16(false, (a), false, (b), (short)0, (c), false, false)

// A/B fragment from row-major memory: lane&15 selects the row, per-lane
// columns are two contiguous 8-element runs at half8 and 16+half8.
// All call sites are 16-byte aligned -> vector loads.
template <typename T>
__device__ __forceinline__ vh16 frag_row(const T* __restrict__ p, int half8) {
  vh16 r;
  if constexpr (sizeof(T) == 2) {
    vh8 lo = *(const vh8*)(p + half8);
    vh8 hi = *(const vh8*)(p + 16 + half8);
#pragma unroll
    for (int j = 0; j < 8; ++j) { r[j] = lo[j]; r[8 + j] = hi[j]; }
  } else {
    vf4 a0 = *(const vf4*)(p + half8);
    vf4 a1 = *(const vf4*)(p + half8 + 4);
    vf4 b0 = *(const vf4*)(p + 16 + half8);
    vf4 b1 = *(const vf4*)(p + 16 + half8 + 4);
#pragma unroll
    for (int j = 0; j < 4; ++j) {
      r[j]      = (_Float16)a0[j];
      r[4 + j]  = (_Float16)a1[j];
      r[8 + j]  = (_Float16)b0[j];
      r[12 + j] = (_Float16)b1[j];
    }
  }
  return r;
}

// ---------------------------------------------------------------------------
// WMMA GEMM: out[M,256] = act(A[M,256] @ W^T + bias [+ residual]) with
// optional per-row gate scale + accumulate (MoE combine).  W is f16 [N,K]
// row-major.  Grid: (M/64, 4), block 128 (4 waves, 16x64 strip per wave).
// ---------------------------------------------------------------------------
template <typename AT>
__global__ __launch_bounds__(128) void gemm16(
    const AT* __restrict__ A,
    const _Float16* __restrict__ W,
    const float* __restrict__ bias,
    float* __restrict__ outF, _Float16* __restrict__ outH,
    const float* __restrict__ residual,
    const float* __restrict__ rowScale, int applyScale,
    float* __restrict__ accum, int relu) {
  const int lane  = threadIdx.x & 31;
  const int wave  = threadIdx.x >> 5;
  const int half8 = (lane >> 4) * 8;
  const int mBase = blockIdx.x * 64 + wave * 16;
  const int nBase = blockIdx.y * 64;
  const int mrow  = mBase + (lane & 15);

  if (rowScale) {  // MoE: skip tiles where no token routed to this expert
    float s = __builtin_fabsf(rowScale[(size_t)mrow * Ee]);
#pragma unroll
    for (int o = 16; o > 0; o >>= 1) s += __shfl_xor(s, o, 32);
    if (s == 0.f) return;  // wave-uniform
  }

  vf8 acc[4];
#pragma unroll
  for (int t = 0; t < 4; ++t) acc[t] = vf8{};

  const AT* ap = A + (size_t)mrow * Hh;
  const _Float16* wp = W + (size_t)(nBase + (lane & 15)) * Hh;
#pragma unroll
  for (int k0 = 0; k0 < Hh; k0 += 32) {
    vh16 a = frag_row(ap + k0, half8);
#pragma unroll
    for (int t = 0; t < 4; ++t)
      acc[t] = WMMA16(a, frag_row(wp + (size_t)t * 16 * Hh + k0, half8), acc[t]);
  }

#pragma unroll
  for (int t = 0; t < 4; ++t) {
#pragma unroll
    for (int r = 0; r < 8; ++r) {
      int m = mBase + r + (lane >> 4) * 8;
      int n = nBase + t * 16 + (lane & 15);
      float v = acc[t][r] + bias[n];
      if (residual) v += residual[(size_t)m * Hh + n];
      if (relu) v = v > 0.f ? v : 0.f;
      if (accum) {
        float sc = applyScale ? rowScale[(size_t)m * Ee] : 1.f;
        accum[(size_t)m * Hh + n] += sc * v;
      } else {
        if (outF) outF[(size_t)m * Hh + n] = v;
        if (outH) outH[(size_t)m * Hh + n] = (_Float16)v;
      }
    }
  }
}

// ---------------------------------------------------------------------------
// Weight prep: fp32 -> f16 copy, and LDS-tiled transpose+convert for the
// expert weights (stored [E][K][N], needed as [E][N][K]).
// ---------------------------------------------------------------------------
__global__ void cvt_f16_kernel(const float* __restrict__ in, _Float16* __restrict__ out, int n) {
  int i = blockIdx.x * 256 + threadIdx.x;
  if (i < n) out[i] = (_Float16)in[i];
}

__global__ void transpose_cvt_kernel(const float* __restrict__ in, _Float16* __restrict__ outT) {
  __shared__ float t[16][17];
  int e = blockIdx.z;
  int n0 = blockIdx.x * 16, k0 = blockIdx.y * 16;
  int tx = threadIdx.x & 15, ty = threadIdx.x >> 4;   // 256 threads = 16x16
  t[ty][tx] = in[((size_t)e * Hh + (k0 + ty)) * Hh + n0 + tx];
  __syncthreads();
  outT[((size_t)e * Hh + (n0 + ty)) * Hh + k0 + tx] = (_Float16)t[tx][ty];
}

// ---------------------------------------------------------------------------
// x = item_emb[id]*sqrt(H) + pos_emb[pos],  pos = (l+1) if id!=0 else 0
// ---------------------------------------------------------------------------
__global__ void embed_kernel(const int* __restrict__ logs,
                             const float* __restrict__ iemb,
                             const float* __restrict__ pemb,
                             float* __restrict__ x) {
  int bl = blockIdx.x, h = threadIdx.x;
  int id = logs[bl];
  int pos = (id != 0) ? (bl % Ll) + 1 : 0;
  x[(size_t)bl * Hh + h] =
      iemb[(size_t)id * Hh + h] * 16.0f + pemb[(size_t)pos * Hh + h];
}

// ---------------------------------------------------------------------------
// LayerNorm over H=256 (optionally of x+x2).  One block (256 thr) per row.
// ---------------------------------------------------------------------------
__global__ void ln_kernel(const float* __restrict__ x, const float* __restrict__ x2,
                          const float* __restrict__ w, const float* __restrict__ b,
                          float eps, float* __restrict__ outF) {
  int row = blockIdx.x, h = threadIdx.x;
  float v = x[(size_t)row * Hh + h];
  if (x2) v += x2[(size_t)row * Hh + h];
  __shared__ float red[Hh];
  red[h] = v; __syncthreads();
  for (int s = Hh / 2; s > 0; s >>= 1) { if (h < s) red[h] += red[h + s]; __syncthreads(); }
  float mean = red[0] / (float)Hh; __syncthreads();
  float d = v - mean;
  red[h] = d * d; __syncthreads();
  for (int s = Hh / 2; s > 0; s >>= 1) { if (h < s) red[h] += red[h + s]; __syncthreads(); }
  float var = red[0] / (float)Hh;
  outF[(size_t)row * Hh + h] = d * rsqrtf(var + eps) * w[h] + b[h];
}

// ---------------------------------------------------------------------------
// Attention: one wave per (query-tile, head, batch). WMMA for S=Q*K^T and P*V.
// ---------------------------------------------------------------------------
__global__ __launch_bounds__(32) void attn_kernel(const _Float16* __restrict__ Q,
                                                  const _Float16* __restrict__ Km,
                                                  const _Float16* __restrict__ V,
                                                  _Float16* __restrict__ ctx) {
  const int qt = blockIdx.x, hIdx = blockIdx.y, b = blockIdx.z;
  const int lane = threadIdx.x;
  const int half8 = (lane >> 4) * 8;
  const int hBase = hIdx * HDh;
  const size_t rowbase = (size_t)b * Ll * Hh;

  __shared__ float    sS[16 * LPAD];
  __shared__ _Float16 sP[16 * LPAD];

  // Q fragments (two K-steps over head dim 64)
  int qrow = qt * 16 + (lane & 15);
  int qr = qrow < Ll ? qrow : Ll - 1;
  const _Float16* qp = Q + rowbase + (size_t)qr * Hh + hBase;
  vh16 a0 = frag_row(qp, half8);
  vh16 a1 = frag_row(qp + 32, half8);

  // Scores with causal mask -> LDS
  for (int kt = 0; kt < 13; ++kt) {
    int krow = kt * 16 + (lane & 15);
    int kr = krow < Ll ? krow : Ll - 1;
    const _Float16* kp = Km + rowbase + (size_t)kr * Hh + hBase;
    vf8 c = vf8{};
    c = WMMA16(a0, frag_row(kp, half8), c);
    c = WMMA16(a1, frag_row(kp + 32, half8), c);
#pragma unroll
    for (int r = 0; r < 8; ++r) {
      int m = r + (lane >> 4) * 8;     // query within tile
      int n = lane & 15;               // key within tile
      int qg = qt * 16 + m, kg = kt * 16 + n;
      float s = c[r] * 0.125f;         // 1/sqrt(64)
      if (kg > qg || kg >= Ll || qg >= Ll) s = -__builtin_inff();
      sS[m * LPAD + kt * 16 + n] = s;
    }
  }
  // zero probability padding (cols 208..223)
  for (int idx = lane; idx < 16 * (LPAD - 208); idx += 32) {
    int m = idx / (LPAD - 208), n = 208 + idx % (LPAD - 208);
    sP[m * LPAD + n] = (_Float16)0.f;
  }
  __syncthreads();

  // softmax (two pass) per query row
  for (int m = 0; m < 16; ++m) {
    float mx = -__builtin_inff();
    for (int n = lane; n < 208; n += 32) mx = fmaxf(mx, sS[m * LPAD + n]);
#pragma unroll
    for (int o = 16; o > 0; o >>= 1) mx = fmaxf(mx, __shfl_xor(mx, o, 32));
    float sum = 0.f;
    for (int n = lane; n < 208; n += 32) {
      float s = sS[m * LPAD + n];
      float e = (s == -__builtin_inff()) ? 0.f : __expf(s - mx);
      sP[m * LPAD + n] = (_Float16)e;
      sum += e;
    }
#pragma unroll
    for (int o = 16; o > 0; o >>= 1) sum += __shfl_xor(sum, o, 32);
    float inv = sum > 0.f ? 1.f / sum : 0.f;
    for (int n = lane; n < 208; n += 32)
      sP[m * LPAD + n] = (_Float16)((float)sP[m * LPAD + n] * inv);
  }
  __syncthreads();

  // ctx = P @ V   (B operand gathered column-major from V[L, H])
#pragma unroll
  for (int dt = 0; dt < 4; ++dt) {
    vf8 c = vf8{};
    int col = hBase + dt * 16 + (lane & 15);
    for (int kk = 0; kk < LPAD; kk += 32) {
      vh16 a = frag_row(sP + (lane & 15) * LPAD + kk, half8);
      vh16 bf;
#pragma unroll
      for (int j = 0; j < 8; ++j) {
        int kA = kk + half8 + j;       if (kA > Ll - 1) kA = Ll - 1;
        int kB = kk + 16 + half8 + j;  if (kB > Ll - 1) kB = Ll - 1;
        bf[j]     = V[rowbase + (size_t)kA * Hh + col];
        bf[8 + j] = V[rowbase + (size_t)kB * Hh + col];
      }
      c = WMMA16(a, bf, c);
    }
#pragma unroll
    for (int r = 0; r < 8; ++r) {
      int m = r + (lane >> 4) * 8;
      int qg = qt * 16 + m;
      if (qg < Ll)
        ctx[rowbase + (size_t)qg * Hh + hBase + dt * 16 + (lane & 15)] = (_Float16)c[r];
    }
  }
}

// ---------------------------------------------------------------------------
// Gate: gl[e] = x . gate_w[e] + gb[e]; top-2 softmax -> comb[M,8]
// ---------------------------------------------------------------------------
__global__ void gate_kernel(const float* __restrict__ x, const float* __restrict__ gw,
                            const float* __restrict__ gb, float* __restrict__ comb) {
  int row = blockIdx.x;
  int e = threadIdx.x >> 5, lane = threadIdx.x & 31;
  const float* xp = x + (size_t)row * Hh;
  const float* wp = gw + (size_t)e * Hh;
  float s = 0.f;
  for (int h = lane; h < Hh; h += 32) s += xp[h] * wp[h];
#pragma unroll
  for (int o = 16; o > 0; o >>= 1) s += __shfl_xor(s, o, 32);
  __shared__ float gl[Ee];
  if (lane == 0) gl[e] = s + gb[e];
  __syncthreads();
  if (threadIdx.x == 0) {
    int i0 = 0; float v0 = gl[0];
    for (int j = 1; j < Ee; ++j) if (gl[j] > v0) { v0 = gl[j]; i0 = j; }
    int i1 = -1; float v1 = -__builtin_inff();
    for (int j = 0; j < Ee; ++j) { if (j == i0) continue; if (gl[j] > v1) { v1 = gl[j]; i1 = j; } }
    float e1 = __expf(v1 - v0);   // v0 is the max
    float inv = 1.f / (1.f + e1);
    float c[Ee];
#pragma unroll
    for (int j = 0; j < Ee; ++j) c[j] = 0.f;
    c[i0] = inv; c[i1] = e1 * inv;
    for (int j = 0; j < Ee; ++j) comb[(size_t)row * Ee + j] = c[j];
  }
}

// ---------------------------------------------------------------------------
// Final: LN(last row), + user_emb, dot item_emb -> out[b]
// ---------------------------------------------------------------------------
__global__ void final_kernel(const float* __restrict__ x,
                             const float* __restrict__ lw, const float* __restrict__ lb,
                             const int* __restrict__ uid, const int* __restrict__ iid,
                             const float* __restrict__ uemb, const float* __restrict__ iemb,
                             float* __restrict__ out) {
  int bb = blockIdx.x, h = threadIdx.x;
  size_t row = (size_t)bb * Ll + (Ll - 1);
  float v = x[row * Hh + h];
  __shared__ float red[Hh];
  red[h] = v; __syncthreads();
  for (int s = Hh / 2; s > 0; s >>= 1) { if (h < s) red[h] += red[h + s]; __syncthreads(); }
  float mean = red[0] / (float)Hh; __syncthreads();
  float d = v - mean;
  red[h] = d * d; __syncthreads();
  for (int s = Hh / 2; s > 0; s >>= 1) { if (h < s) red[h] += red[h + s]; __syncthreads(); }
  float var = red[0] / (float)Hh; __syncthreads();
  float y = d * rsqrtf(var + 1e-8f) * lw[h] + lb[h];
  float f = y + uemb[(size_t)uid[bb] * Hh + h];
  red[h] = f * iemb[(size_t)iid[bb] * Hh + h];
  __syncthreads();
  for (int s = Hh / 2; s > 0; s >>= 1) { if (h < s) red[h] += red[h + s]; __syncthreads(); }
  if (h == 0) out[bb] = red[0];
}

// ---------------------------------------------------------------------------
extern "C" void kernel_launch(void* const* d_in, const int* in_sizes, int n_in,
                              void* d_out, int out_size, void* d_ws, size_t ws_size,
                              hipStream_t stream) {
  (void)in_sizes; (void)n_in; (void)out_size; (void)ws_size;
  const int* user_ids   = (const int*)d_in[0];
  const int* log_seqs   = (const int*)d_in[1];
  const int* item_ids   = (const int*)d_in[2];
  const float* item_emb = (const float*)d_in[3];
  const float* pos_emb  = (const float*)d_in[4];
  const float* user_emb = (const float*)d_in[5];
  const float* ln1_w = (const float*)d_in[6];
  const float* ln1_b = (const float*)d_in[7];
  const float* inproj_w = (const float*)d_in[8];
  const float* inproj_b = (const float*)d_in[9];
  const float* outproj_w = (const float*)d_in[10];
  const float* outproj_b = (const float*)d_in[11];
  const float* ln2_w = (const float*)d_in[12];
  const float* ln2_b = (const float*)d_in[13];
  const float* gate_w = (const float*)d_in[14];
  const float* gate_b = (const float*)d_in[15];
  const float* e_w1 = (const float*)d_in[16];
  const float* e_b1 = (const float*)d_in[17];
  const float* e_w2 = (const float*)d_in[18];
  const float* e_b2 = (const float*)d_in[19];
  const float* moeln_w = (const float*)d_in[20];
  const float* moeln_b = (const float*)d_in[21];
  const float* lastln_w = (const float*)d_in[22];
  const float* lastln_b = (const float*)d_in[23];
  float* out = (float*)d_out;

  const size_t M  = (size_t)Bb * Ll;                 // 25600
  const size_t SF = M * Hh * sizeof(float);
  const size_t S2 = M * Hh * sizeof(_Float16);
  const size_t WH = (size_t)Hh * Hh;                 // 65536
  char* p = (char*)d_ws;
  float*    xf    = (float*)p;    p += SF;           // x fp32 / MoE accumulator
  float*    qinf  = (float*)p;    p += SF;           // q_in / ln2-out fp32
  _Float16* q16   = (_Float16*)p; p += S2;           // Q f16 / expert hidden
  _Float16* k16   = (_Float16*)p; p += S2;           // K f16
  _Float16* v16b  = (_Float16*)p; p += S2;           // V f16
  _Float16* c16   = (_Float16*)p; p += S2;           // attention context f16
  float*    comb  = (float*)p;    p += M * Ee * sizeof(float);
  _Float16* wio16 = (_Float16*)p; p += 3 * WH * sizeof(_Float16);   // qkv weights f16
  _Float16* wo16  = (_Float16*)p; p += WH * sizeof(_Float16);       // outproj f16
  _Float16* w1t   = (_Float16*)p; p += Ee * WH * sizeof(_Float16);  // expert w1^T f16
  _Float16* w2t   = (_Float16*)p; p += Ee * WH * sizeof(_Float16);  // expert w2^T f16
  _Float16* h16   = q16;                             // reuse Q buffer for MoE h

  dim3 gG((unsigned)(M / 64), Hh / 64);              // (400, 4)
  dim3 bG(128);
  dim3 gT(Hh / 16, Hh / 16, Ee);                     // transpose tiles

  embed_kernel<<<dim3((unsigned)M), dim3(Hh), 0, stream>>>(log_seqs, item_emb, pos_emb, xf);

  for (int i = 0; i < NBb; ++i) {
    const float* bq = inproj_b + (size_t)i * 3 * Hh;
    const float* bk = bq + Hh;
    const float* bv = bk + Hh;
    const float* bo = outproj_b + (size_t)i * Hh;

    // weight prep (f16 convert; experts also transposed)
    cvt_f16_kernel<<<(unsigned)((3 * WH + 255) / 256), 256, 0, stream>>>(
        inproj_w + (size_t)i * 3 * WH, wio16, (int)(3 * WH));
    cvt_f16_kernel<<<(unsigned)((WH + 255) / 256), 256, 0, stream>>>(
        outproj_w + (size_t)i * WH, wo16, (int)WH);
    transpose_cvt_kernel<<<gT, 256, 0, stream>>>(e_w1 + (size_t)i * Ee * WH, w1t);
    transpose_cvt_kernel<<<gT, 256, 0, stream>>>(e_w2 + (size_t)i * Ee * WH, w2t);

    // q_in = LN1(x)
    ln_kernel<<<(unsigned)M, Hh, 0, stream>>>(xf, nullptr, ln1_w + (size_t)i * Hh,
                                              ln1_b + (size_t)i * Hh, 1e-8f, qinf);
    // Q = q_in @ wq^T + bq ;  K/V from raw x
    gemm16<float><<<gG, bG, 0, stream>>>(qinf, wio16, bq, nullptr, q16,
                                         nullptr, nullptr, 0, nullptr, 0);
    gemm16<float><<<gG, bG, 0, stream>>>(xf, wio16 + WH, bk, nullptr, k16,
                                         nullptr, nullptr, 0, nullptr, 0);
    gemm16<float><<<gG, bG, 0, stream>>>(xf, wio16 + 2 * WH, bv, nullptr, v16b,
                                         nullptr, nullptr, 0, nullptr, 0);
    // attention
    attn_kernel<<<dim3(13, NHh, Bb), 32, 0, stream>>>(q16, k16, v16b, c16);
    // x = q_in + ctx @ wo^T + bo
    gemm16<_Float16><<<gG, bG, 0, stream>>>(c16, wo16, bo, xf, nullptr,
                                            qinf, nullptr, 0, nullptr, 0);
    // x = LN2(x) -> qinf
    ln_kernel<<<(unsigned)M, Hh, 0, stream>>>(xf, nullptr, ln2_w + (size_t)i * Hh,
                                              ln2_b + (size_t)i * Hh, 1e-8f, qinf);
    // gating
    gate_kernel<<<(unsigned)M, 256, 0, stream>>>(qinf, gate_w + (size_t)i * Ee * Hh,
                                                 gate_b + (size_t)i * Ee, comb);
    // MoE: accumulate routed experts into xf
    (void)hipMemsetAsync(xf, 0, SF, stream);
    for (int e = 0; e < Ee; ++e) {
      const float* b1 = e_b1 + ((size_t)i * Ee + e) * Hh;
      const float* b2 = e_b2 + ((size_t)i * Ee + e) * Hh;
      gemm16<float><<<gG, bG, 0, stream>>>(qinf, w1t + (size_t)e * WH, b1, nullptr, h16,
                                           nullptr, comb + e, 0, nullptr, 1);
      gemm16<_Float16><<<gG, bG, 0, stream>>>(h16, w2t + (size_t)e * WH, b2, nullptr, nullptr,
                                              nullptr, comb + e, 1, xf, 0);
    }
    // x = moeLN(ln2_x + moe)
    ln_kernel<<<(unsigned)M, Hh, 0, stream>>>(qinf, xf, moeln_w + (size_t)i * Hh,
                                              moeln_b + (size_t)i * Hh, 1e-5f, xf);
  }

  final_kernel<<<Bb, Hh, 0, stream>>>(xf, lastln_w, lastln_b, user_ids, item_ids,
                                      user_emb, item_emb, out);
}
